// GATELayer_19086834663562
// MI455X (gfx1250) — matrix operand order repaired and checked by reading, not compiled, not run on hardware
//
#include <hip/hip_runtime.h>
#include <hip/hip_bf16.h>
#include <stdint.h>

typedef __bf16 bf16_t;
typedef __attribute__((ext_vector_type(16))) __bf16 v16bf;
typedef __attribute__((ext_vector_type(8)))  float  v8f;

#define GN   8192
#define GIN  512
#define GOUT 256
#define ALPHA 0.2f
#define NEG_INF -9e15f

// -------------------------------------------------------------------------
// Kernel 1: h = x @ W  (bf16 WMMA, f32 accumulate)
// One wave -> one 16x16 tile of h. Also emits hB: bf16 copy of h pre-swizzled
// into the WMMA B-operand layout for kernel 3 (per-lane-contiguous 32B).
// -------------------------------------------------------------------------
__global__ __launch_bounds__(128) void k_gemm_h(
    const float* __restrict__ x, const float* __restrict__ W,
    float* __restrict__ h, bf16_t* __restrict__ hB)
{
  const int wave = threadIdx.x >> 5;
  const int lane = threadIdx.x & 31;
  const int half = lane >> 4;
  const int l16  = lane & 15;
  const int r0 = blockIdx.y * 16;                    // node rows
  const int c0 = (blockIdx.x * 4 + wave) * 16;       // out cols

  v8f acc = {};
  for (int k = 0; k < GIN; k += 32) {
    v16bf a, b;
    // A operand: x[16x32] tile, documented 16-bit A layout
#pragma unroll
    for (int v = 0; v < 8; ++v) {
      int kk = ((v < 4) ? 2 * v : 16 + 2 * (v - 4)) + half * 8;
      float f0 = x[(size_t)(r0 + l16) * GIN + k + kk];
      float f1 = x[(size_t)(r0 + l16) * GIN + k + kk + 1];
      a[2 * v]     = (__bf16)f0;
      a[2 * v + 1] = (__bf16)f1;
    }
    // B operand: W[32x16] tile; lane col = l16, K group by lane half
#pragma unroll
    for (int v = 0; v < 8; ++v) {
      int kk = ((v < 4) ? 2 * v : 16 + 2 * (v - 4)) + half * 8;
      float f0 = W[(size_t)(k + kk)     * GOUT + c0 + l16];
      float f1 = W[(size_t)(k + kk + 1) * GOUT + c0 + l16];
      b[2 * v]     = (__bf16)f0;
      b[2 * v + 1] = (__bf16)f1;
    }
    acc = __builtin_amdgcn_wmma_f32_16x16x32_bf16(
        false, a, false, b, (short)0, acc, false, false);
  }

  // C layout: lanes 0-15 N=lane, M=v ; lanes 16-31 N=lane-16, M=v+8
#pragma unroll
  for (int v = 0; v < 8; ++v) {
    int m = r0 + v + 8 * half;           // node row
    int c = c0 + l16;                    // out col
    float val = acc[v];
    h[(size_t)m * GOUT + c] = val;
    // swizzled B layout: chunk kc=m/32 contiguous (16 tiles * 512 elems),
    // within tile: lane ((kin/8)&1)*16+col owns 16 bf16, elem (kin&7)+(kin/16)*8
    int kc  = m >> 5;
    int kin = m & 31;
    int nt  = c >> 4;
    int cin = c & 15;
    size_t idx = ((size_t)(kc * 16 + nt)) * 512
               + (size_t)(((kin >> 3) & 1) * 16 + cin) * 16 + (kin & 7) + ((kin >> 4) << 3);
    hB[idx] = (__bf16)val;
  }
}

// -------------------------------------------------------------------------
// Kernel 2: attn_self[i] = h[i,:]@a_self ; attn_neighs[i] = h[i,:]@a_neighs
// One wave per row, shuffle reduction (wave32).
// -------------------------------------------------------------------------
__global__ __launch_bounds__(256) void k_attn_vec(
    const float* __restrict__ h, const float* __restrict__ a_self,
    const float* __restrict__ a_neighs,
    float* __restrict__ s_self, float* __restrict__ s_neigh)
{
  int row  = blockIdx.x * 8 + (threadIdx.x >> 5);
  int lane = threadIdx.x & 31;
  float accS = 0.f, accN = 0.f;
  for (int c = lane; c < GOUT; c += 32) {
    float v = h[(size_t)row * GOUT + c];
    accS += v * a_self[c];
    accN += v * a_neighs[c];
  }
#pragma unroll
  for (int off = 16; off > 0; off >>= 1) {
    accS += __shfl_xor(accS, off, 32);
    accN += __shfl_xor(accN, off, 32);
  }
  if (lane == 0) { s_self[row] = accS; s_neigh[row] = accN; }
}

// -------------------------------------------------------------------------
// Kernel 3: fused masked-softmax (online) + attention@h + ELU.
// 512 blocks x 16 rows; 8 waves/block, each wave owns 32 of 256 out cols.
// h chunks staged LDS via global_load_async_to_lds_b128 (ASYNCcnt).
// -------------------------------------------------------------------------
__global__ __launch_bounds__(256) void k_attn_mm(
    const float* __restrict__ adj, const float* __restrict__ Mm,
    const bf16_t* __restrict__ hB, const float* __restrict__ s_self,
    const float* __restrict__ s_neigh, float* __restrict__ out)
{
  __shared__ float  sc[16][32];        // scores -> p (f32)
  __shared__ __bf16 pbf[16][32];       // p tile, bf16, A-operand friendly
  __shared__ __bf16 hbuf[16 * 512];    // 16 KB h chunk (swizzled B layout)
  __shared__ float  m_s[16], l_s[16], scale_s[16], self_s[16];

  const int tid  = threadIdx.x;
  const int wave = tid >> 5;
  const int lane = tid & 31;
  const int half = lane >> 4;
  const int l16  = lane & 15;
  const int r0   = blockIdx.x * 16;
  const int nt0  = wave * 2;
  const int nt1  = wave * 2 + 1;

  if (tid < 16) {
    m_s[tid] = -__builtin_inff();
    l_s[tid] = 0.f;
    self_s[tid] = s_self[r0 + tid];
  }
  __syncthreads();

  v8f acc0 = {}, acc1 = {};

  const int e  = tid * 2;     // two score elements per thread
  const int ri = e >> 5;      // row in 16-row block
  const int ci = e & 31;      // col in 32-col chunk
  const unsigned lds_dst = (unsigned)(uintptr_t)(&hbuf[tid * 32]); // 64B/thread

  for (int kc = 0; kc < GN / 32; ++kc) {
    const int j0 = kc * 32;

    // ---- async stage of h chunk (contiguous 16KB) into LDS ----
    {
      const bf16_t* src = hB + (size_t)kc * (16 * 512) + tid * 32;
      asm volatile(
          "global_load_async_to_lds_b128 %0, %1, off\n\t"
          "global_load_async_to_lds_b128 %0, %1, off offset:16\n\t"
          "global_load_async_to_lds_b128 %0, %1, off offset:32\n\t"
          "global_load_async_to_lds_b128 %0, %1, off offset:48"
          :: "v"(lds_dst), "v"(src) : "memory");
    }

    // ---- scores: (s_i + n_j) * M, leaky relu, adjacency mask ----
    {
      size_t gidx = (size_t)(r0 + ri) * GN + j0 + ci;
      float2 av = *(const float2*)(adj + gidx);
      float2 mv = *(const float2*)(Mm  + gidx);
      // prefetch next chunk (HBM-bound stream)
      __builtin_prefetch(adj + gidx + 32, 0, 0);
      __builtin_prefetch(Mm  + gidx + 32, 0, 0);
      float base = self_s[ri];
      float a0 = (base + s_neigh[j0 + ci])     * mv.x;
      float a1 = (base + s_neigh[j0 + ci + 1]) * mv.y;
      a0 = a0 > 0.f ? a0 : ALPHA * a0;
      a1 = a1 > 0.f ? a1 : ALPHA * a1;
      sc[ri][ci]     = av.x > 0.f ? a0 : NEG_INF;
      sc[ri][ci + 1] = av.y > 0.f ? a1 : NEG_INF;
    }
    __syncthreads();

    // ---- online max + rescale factor ----
    if (tid < 16) {
      float cm = -__builtin_inff();
#pragma unroll
      for (int j = 0; j < 32; ++j) cm = fmaxf(cm, sc[tid][j]);
      float old = m_s[tid];
      float nm  = fmaxf(old, cm);
      scale_s[tid] = __expf(old - nm);
      m_s[tid] = nm;
    }
    // h chunk must be resident before the barrier below
    asm volatile("s_wait_asynccnt 0" ::: "memory");
    __syncthreads();

    // ---- p = exp(score - m), bf16 pack ----
    {
      float nm = m_s[ri];
      float p0 = __expf(sc[ri][ci] - nm);
      float p1 = __expf(sc[ri][ci + 1] - nm);
      sc[ri][ci] = p0; sc[ri][ci + 1] = p1;
      pbf[ri][ci]     = (__bf16)p0;
      pbf[ri][ci + 1] = (__bf16)p1;
    }
    __syncthreads();

    // ---- denominator update ----
    if (tid < 16) {
      float rs = 0.f;
#pragma unroll
      for (int j = 0; j < 32; ++j) rs += sc[tid][j];
      l_s[tid] = l_s[tid] * scale_s[tid] + rs;
    }

    // ---- rescale accumulators (per-row factor) ----
#pragma unroll
    for (int v = 0; v < 8; ++v) {
      float f = scale_s[v + 8 * half];
      acc0[v] *= f;
      acc1[v] *= f;
    }

    // ---- A operand (P tile) from LDS, documented 16-bit A layout ----
    v16bf amat;
    {
      union { uint4 u[2]; v16bf v; } cvt;
      cvt.u[0] = *(const uint4*)(&pbf[l16][half * 8]);        // K 0..7 (+8)
      cvt.u[1] = *(const uint4*)(&pbf[l16][16 + half * 8]);   // K 16..23 (+8)
      amat = cvt.v;
    }
    // ---- B operands from staged LDS (32B contiguous per lane) ----
    v16bf b0 = *(const v16bf*)(&hbuf[nt0 * 512 + lane * 16]);
    v16bf b1 = *(const v16bf*)(&hbuf[nt1 * 512 + lane * 16]);

    acc0 = __builtin_amdgcn_wmma_f32_16x16x32_bf16(
        false, amat, false, b0, (short)0, acc0, false, false);
    acc1 = __builtin_amdgcn_wmma_f32_16x16x32_bf16(
        false, amat, false, b1, (short)0, acc1, false, false);

    __syncthreads();   // protect sc/pbf/hbuf/l_s for next chunk
  }

  // ---- epilogue: normalize, ELU, store ----
#pragma unroll
  for (int v = 0; v < 8; ++v) {
    int m = r0 + v + 8 * half;
    float invl = 1.0f / l_s[v + 8 * half];
    float o0 = acc0[v] * invl;
    float o1 = acc1[v] * invl;
    o0 = o0 > 0.f ? o0 : __expf(o0) - 1.f;
    o1 = o1 > 0.f ? o1 : __expf(o1) - 1.f;
    out[(size_t)m * GOUT + nt0 * 16 + l16] = o0;
    out[(size_t)m * GOUT + nt1 * 16 + l16] = o1;
  }
}

// -------------------------------------------------------------------------
extern "C" void kernel_launch(void* const* d_in, const int* in_sizes, int n_in,
                              void* d_out, int out_size, void* d_ws, size_t ws_size,
                              hipStream_t stream) {
  (void)in_sizes; (void)n_in; (void)out_size; (void)ws_size;
  const float* x        = (const float*)d_in[0];   // [8192, 512]
  const float* adj      = (const float*)d_in[1];   // [8192, 8192]
  const float* Mm       = (const float*)d_in[2];   // [8192, 8192]
  const float* W        = (const float*)d_in[3];   // [512, 256]
  const float* a_self   = (const float*)d_in[4];   // [256]
  const float* a_neighs = (const float*)d_in[5];   // [256]
  float* out = (float*)d_out;                      // [8192, 256]

  char* ws = (char*)d_ws;
  float*  h      = (float*)ws;                                   // 8 MB
  bf16_t* hB     = (bf16_t*)(ws + (size_t)GN * GOUT * 4);        // 4 MB
  float*  sself  = (float*)(ws + (size_t)GN * GOUT * 6);         // 32 KB
  float*  sneigh = sself + GN;                                   // 32 KB

  // K1: h = x @ W (+ swizzled bf16 copy)
  k_gemm_h<<<dim3(GOUT / 64, GN / 16), 128, 0, stream>>>(x, W, h, hB);
  // K2: attention vectors
  k_attn_vec<<<GN / 8, 256, 0, stream>>>(h, a_self, a_neighs, sself, sneigh);
  // K3: fused online-softmax + attention@h + ELU
  k_attn_mm<<<GN / 16, 256, 0, stream>>>(adj, Mm, hB, sself, sneigh, out);
}
